// LinkPredictionGCN_52493090292250
// MI455X (gfx1250) — matrix operand order repaired and checked by reading, not compiled, not run on hardware
//
#include <hip/hip_runtime.h>
#include <hip/hip_bf16.h>

// ---------------------------------------------------------------------------
// Types for CDNA5 WMMA (wave32): v_wmma_f32_16x16x32_bf16
// ---------------------------------------------------------------------------
typedef __attribute__((ext_vector_type(16))) __bf16 v16bf;
typedef __attribute__((ext_vector_type(8)))  float  v8f;

union BF16x16 { unsigned short u[16]; v16bf v; };

__device__ __forceinline__ unsigned short f2bf_rne(float f) {
    unsigned u = __float_as_uint(f);
    u += 0x7FFFu + ((u >> 16) & 1u);           // round-to-nearest-even
    return (unsigned short)(u >> 16);
}
__device__ __forceinline__ float bf2f(unsigned short h) {
    return __uint_as_float(((unsigned)h) << 16);
}

// ---------------------------------------------------------------------------
// Dense GEMM: C[M x NCOLS] = A[M x K] * B[K x NCOLS], f32 in/out.
// bf16 hi/lo split (3 WMMAs per K-chunk) -> near-fp32 accuracy, memory-bound.
// Block = 256 threads = 8 waves; wave w owns rows [base+16w, base+16w+16),
// iterates all NCOLS/16 column tiles. K processed in chunks of 32.
// EXEC stays all-ones across WMMA (loads clamped/guarded, no divergence).
// ---------------------------------------------------------------------------
template<int NCOLS>
__global__ __launch_bounds__(256)
void gemm_wmma_bf16x2(const float* __restrict__ A,
                      const float* __restrict__ B,
                      float* __restrict__ C,
                      int M, int K) {
    constexpr int NT = NCOLS / 16;
    const int lane = threadIdx.x & 31;
    const int wave = threadIdx.x >> 5;
    const int half = lane >> 4;        // K-half selector for A/B fragments
    const int l16  = lane & 15;

    const int rowBase = blockIdx.x * 128 + wave * 16;
    int r  = rowBase + l16;
    int rc = (r < M) ? r : (M - 1);    // clamp load row; store is guarded
    const float* Arow = A + (size_t)rc * K;

    v8f acc[NT];
    const v8f vzero = {0.f,0.f,0.f,0.f,0.f,0.f,0.f,0.f};
#pragma unroll
    for (int t = 0; t < NT; ++t) acc[t] = vzero;

    for (int k0 = 0; k0 < K; k0 += 32) {
        const bool full = (k0 + 32 <= K);
        // ---- A fragment: lane holds K runs [s1,s1+8) and [s2,s2+8) of row rc
        const int s1 = k0 + 16 * half;
        const int s2 = k0 + 16 * half + 8;
        float af[16];
        if (full) {
            const float4 a0 = *(const float4*)(Arow + s1);
            const float4 a1 = *(const float4*)(Arow + s1 + 4);
            const float4 a2 = *(const float4*)(Arow + s2);
            const float4 a3 = *(const float4*)(Arow + s2 + 4);
            af[0]=a0.x; af[1]=a0.y; af[2]=a0.z; af[3]=a0.w;
            af[4]=a1.x; af[5]=a1.y; af[6]=a1.z; af[7]=a1.w;
            af[8]=a2.x; af[9]=a2.y; af[10]=a2.z; af[11]=a2.w;
            af[12]=a3.x; af[13]=a3.y; af[14]=a3.z; af[15]=a3.w;
        } else {
#pragma unroll
            for (int e = 0; e < 8; ++e) {
                af[e]     = (s1 + e < K) ? Arow[s1 + e] : 0.f;
                af[8 + e] = (s2 + e < K) ? Arow[s2 + e] : 0.f;
            }
        }
        BF16x16 Ahi, Alo;
#pragma unroll
        for (int e = 0; e < 16; ++e) {
            const unsigned short hi = f2bf_rne(af[e]);
            Ahi.u[e] = hi;
            Alo.u[e] = f2bf_rne(af[e] - bf2f(hi));
        }

#pragma unroll
        for (int t = 0; t < NT; ++t) {
            // ---- B fragment: lane holds 16 K-rows at column n
            const int n = t * 16 + l16;
            const float* Bp = B + (size_t)(k0 + 16 * half) * NCOLS + n;
            float bfl[16];
            if (full) {
#pragma unroll
                for (int e = 0; e < 16; ++e) bfl[e] = Bp[(size_t)e * NCOLS];
            } else {
#pragma unroll
                for (int e = 0; e < 16; ++e) {
                    const int kb = k0 + 16 * half + e;
                    bfl[e] = (kb < K) ? Bp[(size_t)e * NCOLS] : 0.f;
                }
            }
            BF16x16 Bhi, Blo;
#pragma unroll
            for (int e = 0; e < 16; ++e) {
                const unsigned short hi = f2bf_rne(bfl[e]);
                Bhi.u[e] = hi;
                Blo.u[e] = f2bf_rne(bfl[e] - bf2f(hi));
            }
            // D = AhiBhi + AloBhi + AhiBlo + C  (drop AloBlo ~2^-16 term)
            acc[t] = __builtin_amdgcn_wmma_f32_16x16x32_bf16(
                         false, Ahi.v, false, Bhi.v, (short)0, acc[t], false, false);
            acc[t] = __builtin_amdgcn_wmma_f32_16x16x32_bf16(
                         false, Alo.v, false, Bhi.v, (short)0, acc[t], false, false);
            acc[t] = __builtin_amdgcn_wmma_f32_16x16x32_bf16(
                         false, Ahi.v, false, Blo.v, (short)0, acc[t], false, false);
        }
    }

    // ---- store: C/D layout -> VGPR i holds row i (lanes 0-15) / 8+i (16-31)
#pragma unroll
    for (int t = 0; t < NT; ++t) {
        const int col = t * 16 + l16;
#pragma unroll
        for (int i = 0; i < 8; ++i) {
            const int row = rowBase + 8 * half + i;
            if (row < M) C[(size_t)row * NCOLS + col] = acc[t][i];
        }
    }
}

// ---------------------------------------------------------------------------
// Degree / normalization kernels
// ---------------------------------------------------------------------------
__global__ void deg_init(float* deg, int N) {
    int i = blockIdx.x * blockDim.x + threadIdx.x;
    if (i < N) deg[i] = 1.0f;                    // self-loop contribution
}
__global__ void deg_accum(const float* /*unused*/, const int* __restrict__ col,
                          float* deg, int E) {
    int e = blockIdx.x * blockDim.x + threadIdx.x;
    if (e < E) unsafeAtomicAdd(&deg[col[e]], 1.0f);
}
__global__ void deg_rsqrt(float* deg, int N) {
    int i = blockIdx.x * blockDim.x + threadIdx.x;
    if (i < N) { float d = deg[i]; deg[i] = (d > 0.f) ? rsqrtf(d) : 0.f; }
}

// ---------------------------------------------------------------------------
// Edge scatter-aggregate: agg[col] += h[row] * dinv[row]*dinv[col]
// Items [0,E) = edges, [E,E+N) = self loops (r==c). One wave per item.
// F = feature width (128: float4/lane, 64: float2/lane) -> coalesced rows.
// ---------------------------------------------------------------------------
template<int F>
__global__ __launch_bounds__(256)
void scatter_agg(const float* __restrict__ h,
                 const int* __restrict__ rows, const int* __restrict__ cols,
                 const float* __restrict__ dinv, float* __restrict__ agg,
                 int E, int N) {
    const int gwave = (int)((blockIdx.x * blockDim.x + threadIdx.x) >> 5);
    const int lane  = threadIdx.x & 31;
    const int total = E + N;
    if (gwave >= total) return;
    int r, c;
    if (gwave < E) { r = rows[gwave]; c = cols[gwave]; }
    else           { r = c = gwave - E; }
    const float nrm = dinv[r] * dinv[c];
    constexpr int V = F / 32;
    const float* src = h   + (size_t)r * F + lane * V;
    float*       dst = agg + (size_t)c * F + lane * V;
    if constexpr (V == 4) {
        float4 v = *(const float4*)src;
        unsafeAtomicAdd(dst + 0, v.x * nrm);
        unsafeAtomicAdd(dst + 1, v.y * nrm);
        unsafeAtomicAdd(dst + 2, v.z * nrm);
        unsafeAtomicAdd(dst + 3, v.w * nrm);
    } else {
        float2 v = *(const float2*)src;
        unsafeAtomicAdd(dst + 0, v.x * nrm);
        unsafeAtomicAdd(dst + 1, v.y * nrm);
    }
}

// ---------------------------------------------------------------------------
// In-place bias (+optional ReLU): a[i] += b[i % F]  (F power of two)
// ---------------------------------------------------------------------------
__global__ void bias_act(float* __restrict__ a, const float* __restrict__ b,
                         size_t total, int fmask, int relu) {
    size_t i = (size_t)blockIdx.x * blockDim.x + threadIdx.x;
    if (i < total) {
        float v = a[i] + b[(int)(i & fmask)];
        if (relu) v = fmaxf(v, 0.f);
        a[i] = v;
    }
}

// ---------------------------------------------------------------------------
// Decode: scores[e] = dot(z[ea[e]], z[eb[e]]) over 64 dims. One wave / edge.
// ---------------------------------------------------------------------------
__global__ __launch_bounds__(256)
void decode_dot(const float* __restrict__ z,
                const int* __restrict__ ea, const int* __restrict__ eb,
                float* __restrict__ out, int E) {
    const int gwave = (int)((blockIdx.x * blockDim.x + threadIdx.x) >> 5);
    const int lane  = threadIdx.x & 31;
    if (gwave >= E) return;
    const int a = ea[gwave], b = eb[gwave];
    const float2 za = *(const float2*)(z + (size_t)a * 64 + lane * 2);
    const float2 zb = *(const float2*)(z + (size_t)b * 64 + lane * 2);
    float s = za.x * zb.x + za.y * zb.y;
#pragma unroll
    for (int off = 16; off > 0; off >>= 1) s += __shfl_xor(s, off, 32);
    if (lane == 0) out[gwave] = s;
}

// ---------------------------------------------------------------------------
// Host launcher
// ---------------------------------------------------------------------------
extern "C" void kernel_launch(void* const* d_in, const int* in_sizes, int n_in,
                              void* d_out, int out_size, void* d_ws, size_t ws_size,
                              hipStream_t stream) {
    const float* x  = (const float*)d_in[0];   // [N,500]
    const float* W1 = (const float*)d_in[1];   // [500,128]
    const float* b1 = (const float*)d_in[2];   // [128]
    const float* W2 = (const float*)d_in[3];   // [128,64]
    const float* b2 = (const float*)d_in[4];   // [64]
    const int*   ei = (const int*)d_in[5];     // [2,E]

    const int N = in_sizes[0] / 500;
    const int E = in_sizes[5] / 2;
    const int* erow = ei;        // edge_index[0]
    const int* ecol = ei + E;    // edge_index[1]

    // Workspace layout (floats). h region is dead after scatter-1, so the
    // layer-2 GEMM output (g2) and aggregate (agg2) overlay it.
    float* ws = (float*)d_ws;
    const size_t hOff    = ((size_t)N + 1023) & ~(size_t)1023;
    const size_t agg1Off = hOff + (size_t)N * 128;
    float* deg  = ws;                       // [N] -> becomes dinv in place
    float* h    = ws + hOff;                // [N,128]
    float* agg1 = ws + agg1Off;             // [N,128] -> hrelu in place
    float* g2   = h;                        // [N,64]
    float* agg2 = h + (size_t)N * 64;       // [N,64] -> z in place

    const int T = 256;
    // zero layer-1 aggregate (atomics accumulate)
    hipMemsetAsync(agg1, 0, (size_t)N * 128 * sizeof(float), stream);

    // --- normalization coefficients
    deg_init  <<<(N + T - 1) / T, T, 0, stream>>>(deg, N);
    deg_accum <<<(E + T - 1) / T, T, 0, stream>>>(nullptr, ecol, deg, E);
    deg_rsqrt <<<(N + T - 1) / T, T, 0, stream>>>(deg, N);

    // --- layer 1: h = x @ W1 ; agg1 = scatter(h) ; hrelu = relu(agg1+b1)
    const int gblocks = (N + 127) / 128;
    gemm_wmma_bf16x2<128><<<gblocks, T, 0, stream>>>(x, W1, h, N, 500);
    {
        const int total = E + N;
        scatter_agg<128><<<(total + 7) / 8, T, 0, stream>>>(h, erow, ecol, deg,
                                                            agg1, E, N);
    }
    bias_act<<<(int)(((size_t)N * 128 + T - 1) / T), T, 0, stream>>>(
        agg1, b1, (size_t)N * 128, 127, 1);

    // --- layer 2: g2 = hrelu @ W2 ; agg2 = scatter(g2) ; z = agg2 + b2
    hipMemsetAsync(agg2, 0, (size_t)N * 64 * sizeof(float), stream);
    gemm_wmma_bf16x2<64><<<gblocks, T, 0, stream>>>(agg1, W2, g2, N, 128);
    {
        const int total = E + N;
        scatter_agg<64><<<(total + 7) / 8, T, 0, stream>>>(g2, erow, ecol, deg,
                                                           agg2, E, N);
    }
    bias_act<<<(int)(((size_t)N * 64 + T - 1) / T), T, 0, stream>>>(
        agg2, b2, (size_t)N * 64, 63, 0);

    // --- decode: per-edge dot of endpoint embeddings
    decode_dot<<<(E + 7) / 8, T, 0, stream>>>(agg2, erow, ecol,
                                              (float*)d_out, E);
}
